// Attention_386547057357
// MI455X (gfx1250) — matrix operand order repaired and checked by reading, not compiled, not run on hardware
//
#include <hip/hip_runtime.h>
#include <hip/hip_bf16.h>

typedef __attribute__((ext_vector_type(16))) _Float16 v16h;
typedef __attribute__((ext_vector_type(8)))  _Float16 h8;
typedef __attribute__((ext_vector_type(8)))  float    v8f;
typedef __attribute__((ext_vector_type(4)))  float    f4;

#define DIM    512
#define NHEAD  8
#define DHEAD  64
#define INNER  512
#define CTX    1024
#define NOCT   11
#define BATCH  8
#define SEQ    256
#define MEMN   768
#define TOTAL  1024

// ---------------------------------------------------------------------------
// WMMA helpers (CDNA5 f16 16x16x32, fp32 accumulate)
// ---------------------------------------------------------------------------
__device__ __forceinline__ v8f wmma_f16(v16h a, v16h b, v8f c) {
  return __builtin_amdgcn_wmma_f32_16x16x32_f16(false, a, false, b, (short)0, c,
                                                false, false);
}

// A fragment (16x32, MxK): lane holds row M=lane&15; K=(e&7)+8*kh+16*(e>>3)
__device__ __forceinline__ v16h fragA(const _Float16* rowp, int kh) {
  v16h f;
#pragma unroll
  for (int e = 0; e < 16; ++e) f[e] = rowp[(e & 7) + 8 * kh + 16 * (e >> 3)];
  return f;
}
// B fragment (32x16, KxN) from LDS stored transposed [n][k]: K=(e&7)+8*kh+16*(e>>3)
__device__ __forceinline__ v16h fragB(const _Float16* rowp, int kh) {
  v16h f;
#pragma unroll
  for (int e = 0; e < 16; ++e) f[e] = rowp[(e & 7) + 8 * kh + 16 * (e >> 3)];
  return f;
}
// pack two h8 (contiguous runs) into an A fragment
__device__ __forceinline__ v16h fragA_from(h8 lo, h8 hi) {
  v16h f;
#pragma unroll
  for (int e = 0; e < 8; ++e) { f[e] = lo[e]; f[8 + e] = hi[e]; }
  return f;
}
__device__ __forceinline__ h8 cvt8(f4 a, f4 b) {
  h8 r;
#pragma unroll
  for (int e = 0; e < 4; ++e) { r[e] = (_Float16)a[e]; r[4 + e] = (_Float16)b[e]; }
  return r;
}

// ---------------------------------------------------------------------------
// Kernel 1: RW^T[i][s] = sum_o PE(s,o) * W_rel[o,i]   (i = h*64+d, 512x1024)
// ---------------------------------------------------------------------------
__global__ __launch_bounds__(256) void rw_kernel(const float* __restrict__ W_rel,
                                                 _Float16* __restrict__ RWT) {
  int t = blockIdx.x * 256 + threadIdx.x;  // 512*1024 threads
  int s = t & (CTX - 1);
  int i = t >> 10;
  float fs = (float)s;
  float acc = 0.f;
#pragma unroll
  for (int o = 0; o < NOCT; ++o) {
    float mult = 3.14159265358979f * (float)(1 << o) * (1.0f / 1024.0f);
    float sc = fs * mult;
    acc += __sinf(sc) * W_rel[o * INNER + i] +
           __cosf(sc) * W_rel[(NOCT + o) * INNER + i];
  }
  RWT[(size_t)i * CTX + s] = (_Float16)acc;
}

// ---------------------------------------------------------------------------
// Kernel 2: qkv GEMM: concat(memory,x)(1024x512) @ W_qkv(512x1536) per batch.
// Epilogue selectors (q/k/v, head, mem-boundary) are wave/block-uniform.
// ---------------------------------------------------------------------------
__global__ __launch_bounds__(256) void qkv_kernel(
    const float* __restrict__ x, const float* __restrict__ memory,
    const float* __restrict__ W_qkv, const float* __restrict__ u_emb,
    const float* __restrict__ v_emb, _Float16* __restrict__ q_u,
    _Float16* __restrict__ q_v, _Float16* __restrict__ kmat,
    _Float16* __restrict__ vmat) {
  __shared__ _Float16 As[64][40];
  __shared__ _Float16 Bs[128][40];
  int b    = blockIdx.z;
  int row0 = blockIdx.y * 64;
  int col0 = blockIdx.x * 128;
  int tid = threadIdx.x, lane = tid & 31, wave = tid >> 5;
  int lrow = lane & 15, kh = lane >> 4;
  int wr = (wave >> 1) * 16, wc = (wave & 1) * 64;

  int ar = tid >> 2, ak = (tid & 3) * 8;   // A: row, 8 contiguous k
  int bk = tid >> 3, bn = (tid & 7) * 16;  // B: k, 16 contiguous n
  int arg = row0 + ar;
  const float* arow = (arg < MEMN)
                          ? &memory[((size_t)b * MEMN + arg) * DIM]
                          : &x[((size_t)b * SEQ + (arg - MEMN)) * DIM];

  v8f acc[4] = {};
  for (int k0 = 0; k0 < DIM; k0 += 32) {
    const f4* ap = (const f4*)(arow + k0 + ak);
    f4 a0 = ap[0], a1 = ap[1];
    const f4* wp = (const f4*)&W_qkv[(size_t)(k0 + bk) * 1536 + col0 + bn];
    f4 w0 = wp[0], w1 = wp[1], w2 = wp[2], w3 = wp[3];
    *(h8*)&As[ar][ak] = cvt8(a0, a1);
#pragma unroll
    for (int j = 0; j < 4; ++j) {
      Bs[bn + j][bk]      = (_Float16)w0[j];
      Bs[bn + 4 + j][bk]  = (_Float16)w1[j];
      Bs[bn + 8 + j][bk]  = (_Float16)w2[j];
      Bs[bn + 12 + j][bk] = (_Float16)w3[j];
    }
    __syncthreads();
    v16h a = fragA(&As[wr + lrow][0], kh);
#pragma unroll
    for (int t = 0; t < 4; ++t) {
      v16h bb = fragB(&Bs[wc + t * 16 + lrow][0], kh);
      acc[t] = wmma_f16(a, bb, acc[t]);
    }
    __syncthreads();
  }

  // ---- epilogue: all selectors uniform ----
  int cbase = col0 + wc;              // 64-aligned; one (which, head) per wave
  int which = cbase >> 9;             // 0=q 1=k 2=v  (wave-uniform)
  int h     = (cbase & 511) >> 6;     // head (wave-uniform)
  int rbase = row0 + wr;              // 16-aligned

  if (which == 0) {
    if (row0 >= MEMN) {               // block-uniform: whole tile is in x-rows
      size_t base = (((size_t)b * NHEAD + h) * SEQ + (rbase - MEMN)) * DHEAD;
#pragma unroll
      for (int t = 0; t < 4; ++t) {
        int d = t * 16 + lrow;
        float ue = u_emb[d], ve = v_emb[d];
#pragma unroll
        for (int r = 0; r < 8; ++r) {
          size_t o = base + (size_t)(r + 8 * kh) * DHEAD + d;
          q_u[o] = (_Float16)(acc[t][r] + ue);
          q_v[o] = (_Float16)(acc[t][r] + ve);
        }
      }
    }
  } else {
    _Float16* dst = (which == 1) ? kmat : vmat;
    size_t base = (((size_t)b * NHEAD + h) * TOTAL + rbase) * DHEAD;
#pragma unroll
    for (int t = 0; t < 4; ++t) {
      int d = t * 16 + lrow;
#pragma unroll
      for (int r = 0; r < 8; ++r)
        dst[base + (size_t)(r + 8 * kh) * DHEAD + d] = (_Float16)acc[t][r];
    }
  }
}

// ---------------------------------------------------------------------------
// Kernel 3: Qb[bh][n][s] = q_v[bh,n,:] . RWT[h*64+ :][s]   (WMMA GEMM)
// ---------------------------------------------------------------------------
__global__ __launch_bounds__(256) void qb_kernel(const _Float16* __restrict__ q_v,
                                                 const _Float16* __restrict__ RWT,
                                                 _Float16* __restrict__ Qb) {
  __shared__ _Float16 As[64][40];
  __shared__ _Float16 Bs[128][40];
  int bh = blockIdx.z;
  int h = bh & 7;
  int row0 = blockIdx.y * 64;   // n
  int col0 = blockIdx.x * 128;  // s
  int tid = threadIdx.x, lane = tid & 31, wave = tid >> 5;
  int lrow = lane & 15, kh = lane >> 4;
  int wr = (wave >> 1) * 16, wc = (wave & 1) * 64;
  int ar = tid >> 2, ak = (tid & 3) * 8;
  int bk = tid >> 3, bn = (tid & 7) * 16;
  v8f acc[4] = {};
  for (int k0 = 0; k0 < DHEAD; k0 += 32) {
    h8 av = *(const h8*)&q_v[((size_t)bh * SEQ + row0 + ar) * DHEAD + k0 + ak];
    const h8* rp = (const h8*)&RWT[(size_t)(h * DHEAD + k0 + bk) * CTX + col0 + bn];
    h8 r0 = rp[0], r1 = rp[1];
    *(h8*)&As[ar][ak] = av;
#pragma unroll
    for (int j = 0; j < 8; ++j) {
      Bs[bn + j][bk]     = r0[j];
      Bs[bn + 8 + j][bk] = r1[j];
    }
    __syncthreads();
    v16h a = fragA(&As[wr + lrow][0], kh);
#pragma unroll
    for (int t = 0; t < 4; ++t) {
      v16h bb = fragB(&Bs[wc + t * 16 + lrow][0], kh);
      acc[t] = wmma_f16(a, bb, acc[t]);
    }
    __syncthreads();
  }
  size_t base = ((size_t)bh * SEQ + row0 + wr) * CTX + col0 + wc;
#pragma unroll
  for (int t = 0; t < 4; ++t)
#pragma unroll
    for (int r = 0; r < 8; ++r)
      Qb[base + (size_t)(r + 8 * kh) * CTX + t * 16 + lrow] = (_Float16)acc[t][r];
}

// ---------------------------------------------------------------------------
// Kernel 4: flash attention. Block = 4 waves x 16 query rows.
// ---------------------------------------------------------------------------
__global__ __launch_bounds__(128) void attn_kernel(
    const _Float16* __restrict__ q_u, const _Float16* __restrict__ kmat,
    const _Float16* __restrict__ vmat, const _Float16* __restrict__ Qb,
    _Float16* __restrict__ ctx) {
  __shared__ _Float16 kts[64][72];     // [m_local][d]
  __shared__ _Float16 vts[64][72];     // [d][m_local]
  __shared__ _Float16 pls[4][16][72];  // per-wave P staging
  int bh = blockIdx.y;
  int b = bh >> 3, h = bh & 7;
  int tid = threadIdx.x, lane = tid & 31, wave = tid >> 5;
  int lrow = lane & 15, kh = lane >> 4;
  int nbase = blockIdx.x * 64 + wave * 16;

  // q fragments: contiguous 8-half runs -> vector global loads
  const _Float16* qp = q_u + ((size_t)bh * SEQ + nbase + lrow) * DHEAD + 8 * kh;
  v16h qa0 = fragA_from(*(const h8*)(qp), *(const h8*)(qp + 16));
  v16h qa1 = fragA_from(*(const h8*)(qp + 32), *(const h8*)(qp + 48));

  v8f o[4] = {};
  float m_i[8], l_i[8];
#pragma unroll
  for (int r = 0; r < 8; ++r) { m_i[r] = -1e30f; l_i[r] = 0.f; }
  const float scale = 0.125f;  // 64^-0.5
  const _Float16* qbp = Qb + (size_t)bh * SEQ * CTX;
  const _Float16* kbase = kmat + (size_t)bh * TOTAL * DHEAD;
  const _Float16* vbase = vmat + (size_t)bh * TOTAL * DHEAD;

  int ml = tid >> 1, d0 = (tid & 1) * 32;  // coop-load assignment (32 halves)

  for (int mc = 0; mc < TOTAL; mc += 64) {
    __syncthreads();
    {
      const h8* kp = (const h8*)(kbase + (size_t)(mc + ml) * DHEAD + d0);
      const h8* vp = (const h8*)(vbase + (size_t)(mc + ml) * DHEAD + d0);
      h8 ka = kp[0], kb = kp[1], kc = kp[2], kd = kp[3];
      h8 va = vp[0], vb = vp[1], vc = vp[2], vd = vp[3];
      *(h8*)&kts[ml][d0]      = ka;
      *(h8*)&kts[ml][d0 + 8]  = kb;
      *(h8*)&kts[ml][d0 + 16] = kc;
      *(h8*)&kts[ml][d0 + 24] = kd;
#pragma unroll
      for (int j = 0; j < 8; ++j) {
        vts[d0 + j][ml]      = va[j];
        vts[d0 + 8 + j][ml]  = vb[j];
        vts[d0 + 16 + j][ml] = vc[j];
        vts[d0 + 24 + j][ml] = vd[j];
      }
    }
    // prefetch next chunk (global_prefetch_b8), 64B per thread covers 8KB
    if (mc + 64 < TOTAL) {
      const char* pk = (const char*)(kbase + (size_t)(mc + 64) * DHEAD);
      const char* pv = (const char*)(vbase + (size_t)(mc + 64) * DHEAD);
      __builtin_prefetch(pk + tid * 64, 0, 0);
      __builtin_prefetch(pv + tid * 64, 0, 0);
    }
    __syncthreads();

    // scores: S = (q+u) . k^T over d=64 (2 WMMA steps) x 4 m-subtiles
    v8f s[4];
#pragma unroll
    for (int t = 0; t < 4; ++t) {
      v16h b0 = fragB(&kts[t * 16 + lrow][0], kh);
      v16h b1 = fragB(&kts[t * 16 + lrow][32], kh);
      v8f z = {};
      z = wmma_f16(qa0, b0, z);
      s[t] = wmma_f16(qa1, b1, z);
    }

    // + relative term (Qb gather), mask s<0, scale; chunk row-max
    float rm[8];
#pragma unroll
    for (int r = 0; r < 8; ++r) {
      int nq = nbase + r + 8 * kh;
      const _Float16* qrow = qbp + (size_t)nq * CTX;
      float rmax = -1e30f;
#pragma unroll
      for (int t = 0; t < 4; ++t) {
        int m = mc + t * 16 + lrow;
        int sidx = MEMN + nq - m;  // in [-255,1023]; <0 => masked
        float val;
        if (sidx < 0)
          val = -1e30f;
        else
          val = (s[t][r] + (float)qrow[sidx]) * scale;
        s[t][r] = val;
        rmax = fmaxf(rmax, val);
      }
      rm[r] = rmax;
    }
#pragma unroll
    for (int r = 0; r < 8; ++r) {
#pragma unroll
      for (int off = 8; off >= 1; off >>= 1)
        rm[r] = fmaxf(rm[r], __shfl_xor(rm[r], off, 32));
    }

    // online softmax update
    float alpha[8];
#pragma unroll
    for (int r = 0; r < 8; ++r) {
      float nm = fmaxf(m_i[r], rm[r]);
      alpha[r] = __expf(m_i[r] - nm);
      m_i[r] = nm;
      float rs = 0.f;
#pragma unroll
      for (int t = 0; t < 4; ++t) {
        float p = __expf(s[t][r] - nm);
        s[t][r] = p;
        rs += p;
      }
#pragma unroll
      for (int off = 8; off >= 1; off >>= 1) rs += __shfl_xor(rs, off, 32);
      l_i[r] = l_i[r] * alpha[r] + rs;
    }
#pragma unroll
    for (int t = 0; t < 4; ++t)
#pragma unroll
      for (int r = 0; r < 8; ++r) o[t][r] *= alpha[r];

    // stage P (C-layout -> row-major LDS) for A-fragment reload
#pragma unroll
    for (int t = 0; t < 4; ++t)
#pragma unroll
      for (int r = 0; r < 8; ++r)
        pls[wave][r + 8 * kh][t * 16 + lrow] = (_Float16)s[t][r];
    __syncthreads();

    // O += P(16x64) @ V(64x64)
    v16h pa0, pa1;
    const _Float16* pp = &pls[wave][lrow][0];
#pragma unroll
    for (int e = 0; e < 16; ++e) {
      int kidx = (e & 7) + 8 * kh + 16 * (e >> 3);
      pa0[e] = pp[kidx];
      pa1[e] = pp[32 + kidx];
    }
#pragma unroll
    for (int dt = 0; dt < 4; ++dt) {
      v16h vb0 = fragB(&vts[dt * 16 + lrow][0], kh);
      v16h vb1 = fragB(&vts[dt * 16 + lrow][32], kh);
      o[dt] = wmma_f16(pa0, vb0, o[dt]);
      o[dt] = wmma_f16(pa1, vb1, o[dt]);
    }
  }

  // normalize, write context (b, n, h, d) f16
  size_t cbase2 = ((size_t)b * SEQ + nbase) * INNER + h * DHEAD;
#pragma unroll
  for (int r = 0; r < 8; ++r) {
    float inv = 1.0f / l_i[r];
    size_t rb = cbase2 + (size_t)(r + 8 * kh) * INNER;
#pragma unroll
    for (int dt = 0; dt < 4; ++dt)
      ctx[rb + dt * 16 + lrow] = (_Float16)(o[dt][r] * inv);
  }
}

// ---------------------------------------------------------------------------
// Kernel 5: out = context(2048x512) @ W_out(512x512) + b_out   (fp32 output)
// ---------------------------------------------------------------------------
__global__ __launch_bounds__(256) void out_kernel(const _Float16* __restrict__ ctx,
                                                  const float* __restrict__ W_out,
                                                  const float* __restrict__ b_out,
                                                  float* __restrict__ out) {
  __shared__ _Float16 As[64][40];
  __shared__ _Float16 Bs[128][40];
  int row0 = blockIdx.y * 64;
  int col0 = blockIdx.x * 128;
  int tid = threadIdx.x, lane = tid & 31, wave = tid >> 5;
  int lrow = lane & 15, kh = lane >> 4;
  int wr = (wave >> 1) * 16, wc = (wave & 1) * 64;
  int ar = tid >> 2, ak = (tid & 3) * 8;
  int bk = tid >> 3, bn = (tid & 7) * 16;
  v8f acc[4] = {};
  for (int k0 = 0; k0 < INNER; k0 += 32) {
    h8 av = *(const h8*)&ctx[(size_t)(row0 + ar) * INNER + k0 + ak];
    const f4* wp = (const f4*)&W_out[(size_t)(k0 + bk) * DIM + col0 + bn];
    f4 w0 = wp[0], w1 = wp[1], w2 = wp[2], w3 = wp[3];
    *(h8*)&As[ar][ak] = av;
#pragma unroll
    for (int j = 0; j < 4; ++j) {
      Bs[bn + j][bk]      = (_Float16)w0[j];
      Bs[bn + 4 + j][bk]  = (_Float16)w1[j];
      Bs[bn + 8 + j][bk]  = (_Float16)w2[j];
      Bs[bn + 12 + j][bk] = (_Float16)w3[j];
    }
    __syncthreads();
    v16h a = fragA(&As[wr + lrow][0], kh);
#pragma unroll
    for (int t = 0; t < 4; ++t) {
      v16h bb = fragB(&Bs[wc + t * 16 + lrow][0], kh);
      acc[t] = wmma_f16(a, bb, acc[t]);
    }
    __syncthreads();
  }
  size_t base = (size_t)(row0 + wr) * DIM + col0 + wc;
#pragma unroll
  for (int t = 0; t < 4; ++t) {
    float bo = b_out[col0 + wc + t * 16 + lrow];
#pragma unroll
    for (int r = 0; r < 8; ++r)
      out[base + (size_t)(r + 8 * kh) * DIM + t * 16 + lrow] = acc[t][r] + bo;
  }
}

// ---------------------------------------------------------------------------
extern "C" void kernel_launch(void* const* d_in, const int* in_sizes, int n_in,
                              void* d_out, int out_size, void* d_ws,
                              size_t ws_size, hipStream_t stream) {
  (void)in_sizes; (void)n_in; (void)out_size; (void)ws_size;
  const float* x      = (const float*)d_in[0];
  const float* memory = (const float*)d_in[1];
  const float* W_qkv  = (const float*)d_in[2];
  const float* W_rel  = (const float*)d_in[3];
  const float* W_out  = (const float*)d_in[4];
  const float* b_out  = (const float*)d_in[5];
  const float* u_emb  = (const float*)d_in[6];
  const float* v_emb  = (const float*)d_in[7];
  float* out = (float*)d_out;

  char* ws = (char*)d_ws;
  size_t off = 0;
  _Float16* RWT  = (_Float16*)(ws + off); off += (size_t)INNER * CTX * 2;
  _Float16* q_u  = (_Float16*)(ws + off); off += (size_t)BATCH * NHEAD * SEQ * DHEAD * 2;
  _Float16* q_v  = (_Float16*)(ws + off); off += (size_t)BATCH * NHEAD * SEQ * DHEAD * 2;
  _Float16* kmat = (_Float16*)(ws + off); off += (size_t)BATCH * NHEAD * TOTAL * DHEAD * 2;
  _Float16* vmat = (_Float16*)(ws + off); off += (size_t)BATCH * NHEAD * TOTAL * DHEAD * 2;
  _Float16* Qb   = (_Float16*)(ws + off); off += (size_t)BATCH * NHEAD * SEQ * CTX * 2;
  _Float16* ctx  = (_Float16*)(ws + off); off += (size_t)BATCH * SEQ * INNER * 2;

  rw_kernel<<<dim3((INNER * CTX) / 256), 256, 0, stream>>>(W_rel, RWT);
  qkv_kernel<<<dim3(1536 / 128, TOTAL / 64, BATCH), 256, 0, stream>>>(
      x, memory, W_qkv, u_emb, v_emb, q_u, q_v, kmat, vmat);
  qb_kernel<<<dim3(CTX / 128, SEQ / 64, BATCH * NHEAD), 256, 0, stream>>>(q_v, RWT, Qb);
  attn_kernel<<<dim3(SEQ / 64, BATCH * NHEAD), 128, 0, stream>>>(q_u, kmat, vmat, Qb, ctx);
  out_kernel<<<dim3(DIM / 128, (BATCH * SEQ) / 64), 256, 0, stream>>>(ctx, W_out, b_out, out);
}
